// LSTM_attention_71554155151439
// MI455X (gfx1250) — matrix operand order repaired and checked by reading, not compiled, not run on hardware
//
#include <hip/hip_runtime.h>
#include <hip/hip_bf16.h>
#include <math.h>

#define H  512
#define B  32
#define T1 128
#define T2 64
#define V1 32000
#define V2 32000

typedef __attribute__((ext_vector_type(16))) __bf16 v16bf;
typedef __attribute__((ext_vector_type(8)))  float  v8f;
typedef __attribute__((ext_vector_type(8)))  __bf16 bf16x8;
typedef __attribute__((ext_vector_type(8)))  float  f32x8;

__device__ __forceinline__ float sigmoidf_(float x){ return 1.0f/(1.0f+__expf(-x)); }

// monotone float<->uint map for atomicMax on floats
__device__ __forceinline__ unsigned enc_f32(float f){
  int i = __float_as_int(f);
  unsigned u = (unsigned)i;
  return (i < 0) ? ~u : (u | 0x80000000u);
}
__device__ __forceinline__ float dec_f32(unsigned u){
  unsigned v = (u & 0x80000000u) ? (u & 0x7fffffffu) : ~u;
  return __int_as_float((int)v);
}

// ---------------------------------------------------------------------------
// Software grid barrier across co-resident blocks (sense via monotone phase).
// bar[0]=arrive count, bar[1]=phase; both zeroed each call. lp = block-local phase.
// ---------------------------------------------------------------------------
__device__ __forceinline__ void grid_sync(unsigned* bar, int nblk, unsigned* lp){
  __syncthreads();
  if (threadIdx.x == 0){
    unsigned target = ++(*lp);
    if (__hip_atomic_fetch_add(&bar[0], 1u, __ATOMIC_ACQ_REL, __HIP_MEMORY_SCOPE_AGENT)
        == (unsigned)(nblk - 1)){
      __hip_atomic_store(&bar[0], 0u, __ATOMIC_RELAXED, __HIP_MEMORY_SCOPE_AGENT);
      __hip_atomic_store(&bar[1], target, __ATOMIC_RELEASE, __HIP_MEMORY_SCOPE_AGENT);
    } else {
      while (__hip_atomic_load(&bar[1], __ATOMIC_ACQUIRE, __HIP_MEMORY_SCOPE_AGENT) < target)
        __builtin_amdgcn_s_sleep(1);
    }
  }
  __syncthreads();
}

// A-fragment (16x32 bf16, M x K) from row-major bf16 matrix.
// lane l holds row m0+(l&15); K = {8*half+0..7} in e=0..7, {16+8*half+0..7} in e=8..15
__device__ __forceinline__ v16bf load_a_bf16(const __bf16* __restrict__ A, int ld, int m0, int k0){
  int lane = threadIdx.x & 31;
  int half = (lane >> 4) & 1;
  const __bf16* p = A + (size_t)(m0 + (lane & 15)) * ld + k0 + 8*half;
  bf16x8 lo = *(const bf16x8*)p;
  bf16x8 hi = *(const bf16x8*)(p + 16);
  v16bf r;
#pragma unroll
  for (int e=0;e<8;++e){ r[e]=lo[e]; r[e+8]=hi[e]; }
  return r;
}

// Same fragment gathered from f32 source with on-the-fly bf16 convert (ws fallback).
__device__ __forceinline__ v16bf load_a_f32cvt(const float* __restrict__ A, int ld, int m0, int k0){
  int lane = threadIdx.x & 31;
  int half = (lane >> 4) & 1;
  const float* p = A + (size_t)(m0 + (lane & 15)) * ld + k0 + 8*half;
  f32x8 lo = *(const f32x8*)p;
  f32x8 hi = *(const f32x8*)(p + 16);
  v16bf r;
#pragma unroll
  for (int e=0;e<8;++e){ r[e]=(__bf16)lo[e]; r[e+8]=(__bf16)hi[e]; }
  return r;
}

// ---------------------------------------------------------------------------
// One LSTM cell step (device body): temp = U[:,seq_t] + W @ hin ; gates ;
// c update ; write hout (and optionally hs[t]). 8 blocks x 256 threads,
// block owns 64 h-rows across all 4 gate chunks. hin/hout double-buffered
// by the caller (staging reads the full hin; writes go to hout only).
// ---------------------------------------------------------------------------
__device__ __forceinline__ void cell_body(
    const float* __restrict__ Ug, const __bf16* __restrict__ Wb,
    const int* __restrict__ seq, int t, int seqT, int vocab,
    const float* __restrict__ hin, float* __restrict__ hout,
    float* __restrict__ c, float* __restrict__ hs, __bf16* hT)
{
  const int tid = threadIdx.x;
  for (int idx = tid; idx < H*B; idx += 256){
    int k = idx >> 5, b = idx & 31;        // hin is (H,B): hin[k*32+b]
    hT[b*520 + k] = (__bf16)hin[idx];      // h^T bf16, padded rows (bank spread)
  }
  __syncthreads();

  const int lane = tid & 31;
  const int w    = tid >> 5;
  const int mt   = w & 3, nt = w >> 2;     // 4 M-tiles x 2 N-tiles = 8 waves
  const int half = (lane >> 4) & 1;
  const int ln   = lane & 15;
  const int bcol = nt*16 + ln;
  const int hbase = blockIdx.x * 64 + mt * 16;
  const int xb   = seq[bcol * seqT + t];   // seq is (B, T)

  v8f acc[4];                              // i, f, o, g accumulators, C = U[:,x_t]
#pragma unroll
  for (int g=0; g<4; ++g){
    int mrow = g*H + hbase;
#pragma unroll
    for (int j=0;j<8;++j)
      acc[g][j] = Ug[(size_t)(mrow + j + 8*half) * vocab + xb];
  }

  for (int kk = 0; kk < H/32; ++kk){
    int k0 = kk*32;
    // B-fragment (K x N = 32 x 16): lane holds col bcol, K = {8*half+e} in e=0..7, {16+8*half+e} in e=8..15
    const __bf16* bp = &hT[(size_t)bcol*520 + k0 + 8*half];
    bf16x8 blo = *(const bf16x8*)bp;
    bf16x8 bhi = *(const bf16x8*)(bp + 16);
    v16bf bf;
#pragma unroll
    for (int e=0;e<8;++e){ bf[e]=blo[e]; bf[e+8]=bhi[e]; }
#pragma unroll
    for (int g=0; g<4; ++g){
      v16bf af = load_a_bf16(Wb, H, g*H + hbase, k0);
      acc[g] = __builtin_amdgcn_wmma_f32_16x16x32_bf16(
                   false, af, false, bf, (short)0, acc[g], false, false);
    }
  }

#pragma unroll
  for (int j=0;j<8;++j){
    int hidx = hbase + j + 8*half;
    float iv = sigmoidf_(acc[0][j]);
    float fv = sigmoidf_(acc[1][j]);
    float ov = sigmoidf_(acc[2][j]);
    float gv = tanhf(acc[3][j]);
    int off = hidx*B + bcol;
    float cn = fv * c[off] + iv * gv;      // c element owned by this lane
    float hn = ov * tanhf(cn);
    c[off] = cn;
    hout[off] = hn;
    if (hs) hs[((size_t)t*B + bcol)*H + hidx] = hn;
  }
}

// ---------------------------------------------------------------------------
// Persistent encoder: all 128 steps in one launch, grid barrier between steps.
// ---------------------------------------------------------------------------
__global__ __launch_bounds__(256) void encoder_persistent(
    const float* __restrict__ U, const __bf16* __restrict__ Wb,
    const int* __restrict__ x,
    float* __restrict__ h0, float* __restrict__ h1,
    float* __restrict__ c, float* __restrict__ hs, unsigned* __restrict__ bars)
{
  __shared__ __bf16 hT[B * 520];
  unsigned lp = 0;
  for (int t = 0; t < T1; ++t){
    const float* hin = (t & 1) ? h1 : h0;
    float* hout      = (t & 1) ? h0 : h1;
    cell_body(U, Wb, x, t, T1, V1, hin, hout, c, hs, hT);
    grid_sync(bars, 8, &lp);               // publish hout/c for next step
  }
}

// ---------------------------------------------------------------------------
// Decoder step: LSTM cell -> grid barrier -> attention (4 batches per block),
// write resB[b] = [a, s] bf16, reset per-step softmax accumulators.
// grid = 8 blocks x 256 threads.
// ---------------------------------------------------------------------------
__global__ __launch_bounds__(256) void decoder_step(
    const float* __restrict__ U_, const __bf16* __restrict__ W_b,
    const int* __restrict__ y, int t2,
    const float* __restrict__ hin, float* __restrict__ hout,
    float* __restrict__ c, const float* __restrict__ hs,
    __bf16* __restrict__ resB, unsigned* __restrict__ wsMax,
    float* __restrict__ wsSum, unsigned* __restrict__ bar)
{
  __shared__ __bf16 hT[B * 520];
  __shared__ float sS[H];
  __shared__ float eL[T1];
  __shared__ float redM, redS;
  unsigned lp = 0;

  cell_body(U_, W_b, y, t2, T2, V2, hin, hout, c, nullptr, hT);
  grid_sync(bar, 8, &lp);                  // s (=hout) complete across blocks

  const int tid = threadIdx.x;
  const int w = tid >> 5, lane = tid & 31;
  for (int bi = 0; bi < 4; ++bi){
    const int b = blockIdx.x * 4 + bi;
    for (int k = tid; k < H; k += 256) sS[k] = hout[k*B + b];
    __syncthreads();

    for (int i = 0; i < 16; ++i){          // 8 waves x 16 t-values
      int t = w*16 + i;
      const float* row = hs + ((size_t)t*B + b)*H;
      float p = 0.f;
#pragma unroll 4
      for (int k = lane; k < H; k += 32) p += row[k]*sS[k];
#pragma unroll
      for (int off=16; off>0; off>>=1) p += __shfl_xor(p, off, 32);
      if (lane == 0) eL[t] = p;
    }
    __syncthreads();
    if (w == 0){
      float m = -3.0e38f;
      for (int i = lane; i < T1; i += 32) m = fmaxf(m, eL[i]);
#pragma unroll
      for (int off=16; off>0; off>>=1) m = fmaxf(m, __shfl_xor(m, off, 32));
      float sum = 0.f;
      for (int i = lane; i < T1; i += 32) sum += __expf(eL[i]-m);
#pragma unroll
      for (int off=16; off>0; off>>=1) sum += __shfl_xor(sum, off, 32);
      if (lane == 0){ redM = m; redS = sum; }
    }
    __syncthreads();
    float m = redM, inv = 1.0f/redS;
    for (int i = tid; i < T1; i += 256) eL[i] = __expf(eL[i]-m)*inv;  // alpha
    __syncthreads();
    for (int hh = tid; hh < H; hh += 256){
      float a = 0.f;
      for (int t = 0; t < T1; ++t) a += hs[((size_t)t*B + b)*H + hh]*eL[t];
      resB[(size_t)b*2*H + hh]     = (__bf16)a;
      resB[(size_t)b*2*H + H + hh] = (__bf16)sS[hh];
    }
    __syncthreads();                        // sS/eL reused next bi
  }
  if (blockIdx.x == 0 && tid < B){ wsMax[tid] = 0u; wsSum[tid] = 0.f; }
}

// ---------------------------------------------------------------------------
// Projection: logits^T (V2 x B) = Vbf (V2 x 2H) @ resB^T ; store logits,
// atomicMax per-batch max. grid = V2/64 blocks x 256 (8 waves, 1 tile each).
// ---------------------------------------------------------------------------
__global__ __launch_bounds__(256) void proj_logits(
    const __bf16* __restrict__ Vb, const float* __restrict__ Vf, int use_bf,
    const __bf16* __restrict__ resB,
    float* __restrict__ out, int t2, unsigned* __restrict__ wsMax)
{
  const int tid = threadIdx.x;
  const int lane = tid & 31;
  const int w    = tid >> 5;
  const int mt   = w & 3, nt = w >> 2;
  const int half = (lane >> 4) & 1;
  const int ln   = lane & 15;
  const int v0   = blockIdx.x*64 + mt*16;
  const int bcol = nt*16 + ln;

  v8f acc = {};
  for (int kk = 0; kk < (2*H)/32; ++kk){
    int k0 = kk*32;
    const __bf16* bp = resB + (size_t)bcol*(2*H) + k0 + 8*half;
    bf16x8 blo = *(const bf16x8*)bp;
    bf16x8 bhi = *(const bf16x8*)(bp + 16);
    v16bf bf;
#pragma unroll
    for (int e=0;e<8;++e){ bf[e]=blo[e]; bf[e+8]=bhi[e]; }
    v16bf af = use_bf ? load_a_bf16(Vb, 2*H, v0, k0)
                      : load_a_f32cvt(Vf, 2*H, v0, k0);
    acc = __builtin_amdgcn_wmma_f32_16x16x32_bf16(
              false, af, false, bf, (short)0, acc, false, false);
  }
  float lm = -3.0e38f;
#pragma unroll
  for (int j=0;j<8;++j){
    int vv = v0 + j + 8*half;
    float lg = acc[j];
    out[((size_t)t2*B + bcol)*V2 + vv] = lg;
    lm = fmaxf(lm, lg);
  }
  atomicMax(&wsMax[bcol], enc_f32(lm));
}

// pass 2: exp(l - max), accumulate per-batch sum.  grid = B*125, block = 256
__global__ __launch_bounds__(256) void softmax_exp(
    float* __restrict__ out, int t2, const unsigned* __restrict__ wsMax,
    float* __restrict__ wsSum)
{
  __shared__ float red[8];
  const int b = blockIdx.x / 125;
  const int v = (blockIdx.x % 125)*256 + threadIdx.x;     // 125*256 == 32000
  size_t idx = ((size_t)t2*B + b)*V2 + v;
  float ex = __expf(out[idx] - dec_f32(wsMax[b]));
  out[idx] = ex;
  float p = ex;
#pragma unroll
  for (int off=16; off>0; off>>=1) p += __shfl_xor(p, off, 32);
  int lane = threadIdx.x & 31, w = threadIdx.x >> 5;
  if (lane == 0) red[w] = p;
  __syncthreads();
  if (threadIdx.x == 0){
    float s = 0.f;
    for (int i=0;i<8;++i) s += red[i];
    atomicAdd(&wsSum[b], s);
  }
}

// pass 3: divide by sum
__global__ __launch_bounds__(256) void softmax_div(
    float* __restrict__ out, int t2, const float* __restrict__ wsSum)
{
  const int b = blockIdx.x / 125;
  const int v = (blockIdx.x % 125)*256 + threadIdx.x;
  size_t idx = ((size_t)t2*B + b)*V2 + v;
  out[idx] = out[idx] * (1.0f / wsSum[b]);
}

__global__ void cvt_f32_bf16(const float* __restrict__ src, __bf16* __restrict__ dst, long long n){
  long long i = (long long)blockIdx.x*256 + threadIdx.x;
  long long stride = (long long)gridDim.x*256;
  for (; i < n; i += stride) dst[i] = (__bf16)src[i];
}
__global__ void zero_f32(float* __restrict__ p, int n){
  int i = blockIdx.x*256 + threadIdx.x;
  if (i < n) p[i] = 0.f;
}

extern "C" void kernel_launch(void* const* d_in, const int* in_sizes, int n_in,
                              void* d_out, int out_size, void* d_ws, size_t ws_size,
                              hipStream_t stream)
{
  (void)in_sizes; (void)n_in; (void)out_size;
  const float* U  = (const float*)d_in[0];   // (4H, V1)
  const float* W  = (const float*)d_in[1];   // (4H, H)
  const float* U_ = (const float*)d_in[2];   // (4H, V2)
  const float* W_ = (const float*)d_in[3];   // (4H, H)
  const float* V  = (const float*)d_in[4];   // (V2, 2H)
  const int*   x  = (const int*)d_in[5];     // (B, T1)
  const int*   y  = (const int*)d_in[6];     // (B, T2)
  float* out = (float*)d_out;                // (T2, B, V2)

  char* ws = (char*)d_ws;
  size_t off = 0;
  auto take = [&](size_t bytes)->void*{
    void* p = ws + off; off += (bytes + 255) & ~(size_t)255; return p;
  };
  // h0,h1,c,bars contiguous so one zero_f32 initializes them all each call
  float*    h0    = (float*)take((size_t)H*B*4);       // state buffer A (H,B)
  float*    h1    = (float*)take((size_t)H*B*4);       // state buffer B (H,B)
  float*    c     = (float*)take((size_t)H*B*4);       // cell state (H,B)
  unsigned* bars  = (unsigned*)take(256*4);            // grid-barrier slots
  __bf16*   resB  = (__bf16*)take((size_t)B*2*H*2);    // [a, s] per batch (B, 2H)
  unsigned* wsMax = (unsigned*)take((size_t)B*4);
  float*    wsSum = (float*)take((size_t)B*4);
  __bf16*   Wb    = (__bf16*)take((size_t)4*H*H*2);
  __bf16*   W_b   = (__bf16*)take((size_t)4*H*H*2);
  float*    hs    = (float*)take((size_t)T1*B*H*4);    // (T1, B, H)
  size_t vbytes   = (size_t)V2*2*H*2;
  int use_vbf = (off + vbytes) <= ws_size;             // bf16 copy of V if ws allows
  __bf16* Vb = use_vbf ? (__bf16*)take(vbytes) : nullptr;

  const int nzero = 3*H*B + 256;                       // h0,h1,c + barrier words
  zero_f32<<<(nzero + 255)/256, 256, 0, stream>>>(h0, nzero);
  cvt_f32_bf16<<<1024, 256, 0, stream>>>(W,  Wb,  (long long)4*H*H);
  cvt_f32_bf16<<<1024, 256, 0, stream>>>(W_, W_b, (long long)4*H*H);
  if (use_vbf)
    cvt_f32_bf16<<<4096, 256, 0, stream>>>(V, Vb, (long long)V2*2*H);

  // encoder: one persistent launch, 128 steps with in-kernel grid barriers
  encoder_persistent<<<8, 256, 0, stream>>>(U, Wb, x, h0, h1, c, hs, bars);
  // after step 127 (odd): final h lives in h0

  // decoder: fused cell+attention, then projection + 2-pass stable softmax
  for (int t2 = 0; t2 < T2; ++t2){
    const float* hin = (t2 & 1) ? h1 : h0;
    float*       hout = (t2 & 1) ? h0 : h1;
    decoder_step<<<8, 256, 0, stream>>>(U_, W_b, y, t2, hin, hout, c, hs,
                                        resB, wsMax, wsSum, bars + 2 + 2*t2);
    proj_logits<<<V2/64, 256, 0, stream>>>(Vb, V, use_vbf, resB, out, t2, wsMax);
    softmax_exp<<<B*125, 256, 0, stream>>>(out, t2, wsMax, wsSum);
    softmax_div<<<B*125, 256, 0, stream>>>(out, t2, wsSum);
  }
}
